// BlockCirculantLinear_87445534146768
// MI455X (gfx1250) — matrix operand order repaired and checked by reading, not compiled, not run on hardware
//
#include <hip/hip_runtime.h>
#include <hip/hip_bf16.h>

// Block-circulant linear as dense f16 WMMA GEMM on gfx1250 (MI455X).
// y[t,n] = sum_k W[n,k] * x[t,k],  W[n,k] = c[n>>6][k>>6][(n-k)&63]
//
// Pipeline:
//   (1) cvt_x_f16 : X f32 -> f16 into ws           (32 MB)
//   (2) expand_w  : c -> dense W f16, [N][K] major (32 MB, L2-resident)
//   (3) bc_gemm   : Y = X * W^T via v_wmma_f32_16x16x32_f16,
//                   tiles staged L2->LDS by the Tensor Data Mover
//                   (tensor_load_to_lds, double-buffered, TENSORcnt-throttled)

typedef _Float16 v8h  __attribute__((ext_vector_type(8)));
typedef _Float16 v16h __attribute__((ext_vector_type(16)));
typedef float    v8f  __attribute__((ext_vector_type(8)));
typedef unsigned int u32x4 __attribute__((ext_vector_type(4)));
typedef unsigned int u32x8 __attribute__((ext_vector_type(8)));

#define M_TOT 4096   // tokens (2*2048)
#define N_TOT 4096   // out features
#define K_TOT 4096   // in features
#define BT    128    // M/N block tile
#define KT    64     // K block tile (two WMMA K-steps)
#define LDSS  72     // padded LDS row stride in halves (64 data + 8 pad = 144B)
#define NK    (K_TOT / KT)

// ---------------------------------------------------------------- (1) convert X
__global__ __launch_bounds__(256) void cvt_x_f16(const float* __restrict__ x,
                                                 _Float16* __restrict__ a) {
  const int g = blockIdx.x * 256 + threadIdx.x;   // one 8-element chunk per thread
  const size_t base = (size_t)g * 8;
  float4 lo = *(const float4*)(x + base);
  float4 hi = *(const float4*)(x + base + 4);
  v8h o;
  o[0] = (_Float16)lo.x; o[1] = (_Float16)lo.y; o[2] = (_Float16)lo.z; o[3] = (_Float16)lo.w;
  o[4] = (_Float16)hi.x; o[5] = (_Float16)hi.y; o[6] = (_Float16)hi.z; o[7] = (_Float16)hi.w;
  *(v8h*)(a + base) = o;
}

// ---------------------------------------------------------------- (2) expand W
// W[n][k] = c[j][i][(bn - bk) & 63], j=n>>6, i=k>>6, bn=n&63, bk=k&63
__global__ __launch_bounds__(256) void expand_w(const float* __restrict__ c,
                                                _Float16* __restrict__ w) {
  const int g  = blockIdx.x * 256 + threadIdx.x;  // one 8-k chunk per thread
  const int n  = g >> 9;                          // K_TOT/8 = 512 chunks per row
  const int k0 = (g & 511) * 8;
  const int i  = k0 >> 6;
  const int bk = k0 & 63;
  const int j  = n >> 6;
  const int bn = n & 63;
  const float* __restrict__ cb = c + ((size_t)j * 64 + i) * 64;
  v8h o;
#pragma unroll
  for (int t = 0; t < 8; ++t)
    o[t] = (_Float16)cb[(unsigned)(bn - bk - t) & 63u];
  *(v8h*)(w + (size_t)n * K_TOT + k0) = o;
}

// ---------------------------------------------------------------- TDM helper
// Build a 2D tensor-DMA descriptor (D# groups 0+1, ISA ch.8) and issue
// tensor_load_to_lds. Tile: 128 rows x 64 halves from a [*][K_TOT] f16 matrix.
// TDM hardware padding inserts 4 dwords (16B) after every 32 dwords (one 128B
// row) so rows land at the 72-half LDS stride the compute waves expect.
__device__ __forceinline__ void tdm_tile_load(unsigned int lds_addr,
                                              unsigned long long gaddr) {
  u32x4 g0;
  g0[0] = 1u;                                        // count=1 valid descriptor
  g0[1] = lds_addr;                                  // LDS byte address
  g0[2] = (unsigned int)gaddr;                       // global_addr[31:0]
  g0[3] = (unsigned int)((gaddr >> 32) & 0x01FFFFFFu) | (2u << 30); // [56:32]|type=2
  u32x8 g1;
  g1[0] = (1u << 17)    // data_size = 2 bytes
        | (1u << 20)    // pad_enable
        | (4u << 22)    // pad_interval code 4 -> every 32 dwords (one 128B row)
        | (3u << 25);   // pad_amount  code 3 -> 4 dwords (16B)
  g1[1] = (K_TOT & 0xFFFFu) << 16;                   // tensor_dim0[15:0] @bits[63:48]
  g1[2] = (K_TOT >> 16) | ((M_TOT & 0xFFFFu) << 16); // dim0 hi | tensor_dim1 lo
  g1[3] = (M_TOT >> 16) | ((unsigned)KT << 16);      // dim1 hi | tile_dim0 = 64
  g1[4] = (unsigned)BT;                              // tile_dim1 = 128, tile_dim2 = 0
  g1[5] = (unsigned)K_TOT;                           // tensor_dim0_stride[31:0]
  g1[6] = 0u;                                        // stride hi | dim1_stride lo
  g1[7] = 0u;
  asm volatile("tensor_load_to_lds %0, %1" :: "s"(g0), "s"(g1) : "memory");
}

// ---------------------------------------------------------------- (3) GEMM
__global__ __launch_bounds__(256) void bc_gemm(const _Float16* __restrict__ A,
                                               const _Float16* __restrict__ W,
                                               const float* __restrict__ bias,
                                               float* __restrict__ Y) {
  __shared__ _Float16 sA[2][BT][LDSS];
  __shared__ _Float16 sB[2][BT][LDSS];

  const int tid  = threadIdx.x;
  const int lane = tid & 31;
  const int wave = tid >> 5;
  const int wr   = wave >> 2;        // 0..1  -> 64-row strip
  const int wc   = wave & 3;         // 0..3  -> 32-col strip
  const int mBase = blockIdx.y * BT;
  const int nBase = blockIdx.x * BT;

  v8f acc[4][2];
#pragma unroll
  for (int mt = 0; mt < 4; ++mt)
#pragma unroll
    for (int nt = 0; nt < 2; ++nt)
      acc[mt][nt] = (v8f){0.f, 0.f, 0.f, 0.f, 0.f, 0.f, 0.f, 0.f};

  const int laneLo = lane & 15;
  const int hi16   = lane >> 4;      // 0 for lanes 0-15, 1 for 16-31

  // Generic pointers to LDS carry the aperture in the high dword; low 32 bits
  // are the wave-relative LDS byte offset the D# wants.
  const unsigned int ldsA0 = (unsigned int)(uintptr_t)&sA[0][0][0];
  const unsigned int ldsA1 = (unsigned int)(uintptr_t)&sA[1][0][0];
  const unsigned int ldsB0 = (unsigned int)(uintptr_t)&sB[0][0][0];
  const unsigned int ldsB1 = (unsigned int)(uintptr_t)&sB[1][0][0];
  const unsigned long long gaA =
      (unsigned long long)(uintptr_t)A + (unsigned long long)mBase * K_TOT * 2ull;
  const unsigned long long gaB =
      (unsigned long long)(uintptr_t)W + (unsigned long long)nBase * K_TOT * 2ull;

  // Prologue: wave 0 kicks off DMA of stage 0.
  if (wave == 0) {
    tdm_tile_load(ldsA0, gaA);
    tdm_tile_load(ldsB0, gaB);
  }

  for (int kt = 0; kt < NK; ++kt) {
    if (wave == 0) {
      if (kt + 1 < NK) {
        // Issue next stage into the other buffer, then wait until only those
        // two DMAs remain outstanding => current stage is complete (in-order).
        const unsigned long long off = (unsigned long long)(kt + 1) * KT * 2ull;
        if ((kt + 1) & 1) { tdm_tile_load(ldsA1, gaA + off); tdm_tile_load(ldsB1, gaB + off); }
        else             { tdm_tile_load(ldsA0, gaA + off); tdm_tile_load(ldsB0, gaB + off); }
        __builtin_amdgcn_s_wait_tensorcnt(2);
      } else {
        __builtin_amdgcn_s_wait_tensorcnt(0);
      }
    }
    __syncthreads();                 // stage kt visible to all 8 waves
    const int buf = kt & 1;

#pragma unroll
    for (int kk = 0; kk < KT; kk += 32) {
      // A fragments: CDNA5 16-bit A layout. Lane l<16: row=l, K 0-7 (v0-3) and
      // 16-23 (v4-7); lane l+16: same row, K 8-15 and 24-31.
      v16h af[4];
#pragma unroll
      for (int mt = 0; mt < 4; ++mt) {
        const int row = wr * 64 + mt * 16 + laneLo;
        const int kb  = kk + hi16 * 8;
        v8h lo = *(const v8h*)&sA[buf][row][kb];
        v8h hh = *(const v8h*)&sA[buf][row][kb + 16];
#pragma unroll
        for (int e = 0; e < 8; ++e) { af[mt][e] = lo[e]; af[mt][e + 8] = hh[e]; }
      }
      // B fragments: lane l<16 holds K 0-7 and 16-23 of column n=l, lanes 16-31
      // hold K 8-15 and 24-31 (matches A). W stored [N][K].
      v16h bf[2];
#pragma unroll
      for (int nt = 0; nt < 2; ++nt) {
        const int row = wc * 32 + nt * 16 + laneLo;
        const int kb  = kk + hi16 * 8;
        v8h lo = *(const v8h*)&sB[buf][row][kb];
        v8h hh = *(const v8h*)&sB[buf][row][kb + 16];
#pragma unroll
        for (int e = 0; e < 8; ++e) { bf[nt][e] = lo[e]; bf[nt][e + 8] = hh[e]; }
      }
#pragma unroll
      for (int mt = 0; mt < 4; ++mt)
#pragma unroll
        for (int nt = 0; nt < 2; ++nt)
          acc[mt][nt] = __builtin_amdgcn_wmma_f32_16x16x32_f16(
              /*neg_a=*/false, af[mt], /*neg_b=*/false, bf[nt],
              /*c_mod=*/(short)0, acc[mt][nt],
              /*reuse_a=*/false, /*reuse_b=*/false);
    }
    __syncthreads();                 // all waves done with buf before TDM reuses it
  }

  // ---- epilogue: C/D layout -> VGPR r: M = r (lanes 0-15) / r+8 (lanes 16-31),
  // N = lane & 15. Add bias, store f32.
  const int rowHi = hi16 * 8;
#pragma unroll
  for (int nt = 0; nt < 2; ++nt) {
    const int col = nBase + wc * 32 + nt * 16 + laneLo;
    const float b = bias[col];
#pragma unroll
    for (int mt = 0; mt < 4; ++mt) {
#pragma unroll
      for (int r = 0; r < 8; ++r) {
        const int row = mBase + wr * 64 + mt * 16 + rowHi + r;
        Y[(size_t)row * N_TOT + col] = acc[mt][nt][r] + b;
      }
    }
  }
}

// ---------------------------------------------------------------- launch
extern "C" void kernel_launch(void* const* d_in, const int* in_sizes, int n_in,
                              void* d_out, int out_size, void* d_ws, size_t ws_size,
                              hipStream_t stream) {
  const float* x    = (const float*)d_in[0];   // [2,2048,4096] f32
  const float* c    = (const float*)d_in[1];   // [64,64,64]    f32
  const float* bias = (const float*)d_in[2];   // [4096]        f32
  float* y = (float*)d_out;                    // [2,2048,4096] f32

  _Float16* aF = (_Float16*)d_ws;                                        // 32 MB
  _Float16* wF = (_Float16*)((char*)d_ws + (size_t)M_TOT * K_TOT * 2);   // 32 MB

  // (1) X -> f16
  cvt_x_f16<<<(M_TOT * K_TOT) / (256 * 8), 256, 0, stream>>>(x, aF);
  // (2) circulant blocks -> dense W f16 [N][K]
  expand_w<<<(N_TOT * (K_TOT / 8)) / 256, 256, 0, stream>>>(c, wF);
  // (3) Y = X * W^T, 128x128 tile per workgroup, TDM-staged k-loop
  dim3 grid(N_TOT / BT, M_TOT / BT);
  bc_gemm<<<grid, 256, 0, stream>>>(aF, wF, bias, y);
}
